// CPPN_28226525069673
// MI455X (gfx1250) — matrix operand order repaired and checked by reading, not compiled, not run on hardware
//
#include <hip/hip_runtime.h>

typedef _Float16 v16h  __attribute__((ext_vector_type(16)));
typedef _Float16 v2h   __attribute__((ext_vector_type(2)));
typedef __fp16   h2raw __attribute__((ext_vector_type(2)));
typedef float    v8f   __attribute__((ext_vector_type(8)));

union Frag {
    v16h v;
    v2h  h[8];
};

__device__ __forceinline__ v2h pk(float a, float b) {
    h2raw r = __builtin_amdgcn_cvt_pkrtz(a, b);   // v_cvt_pk_f16_f32 (RTZ)
    return __builtin_bit_cast(v2h, r);
}

// cross-half exchange (lane ^ 16) of one dword: v_permlanex16_b32 with identity
// lane selects (VALU, no LDS/DS counter) if available, else ds_bpermute fallback.
__device__ __forceinline__ unsigned xdw(unsigned u) {
#if __has_builtin(__builtin_amdgcn_permlanex16)
    return __builtin_amdgcn_permlanex16(u, u, 0x76543210u, 0xfedcba98u, false, false);
#else
    return (unsigned)__shfl_xor((int)u, 16, 32);
#endif
}
__device__ __forceinline__ v2h xh(v2h v) {
    return __builtin_bit_cast(v2h, xdw(__builtin_bit_cast(unsigned, v)));
}
__device__ __forceinline__ float xf(float v) {
    return __builtin_bit_cast(float, xdw(__builtin_bit_cast(unsigned, v)));
}

// tanh: single v_tanh_f32 on gfx1250 if available; else exp2+rcp expansion.
__device__ __forceinline__ float fast_tanh(float x) {
#if __has_builtin(__builtin_amdgcn_tanhf)
    return __builtin_amdgcn_tanhf(x);
#else
    float e = __builtin_amdgcn_exp2f(x * 2.885390081777927f);
    return 1.0f - 2.0f * __builtin_amdgcn_rcpf(e + 1.0f);
#endif
}

__device__ __forceinline__ float fast_sigmoid(float x) {
    float e = __builtin_amdgcn_exp2f(-x * 1.4426950408889634f);
    return __builtin_amdgcn_rcpf(1.0f + e);
}

// CPPN: [N,3] -> Linear(3,16)+tanh -> 8x(Linear(16,16)+tanh) -> Linear(16,1)+sigmoid
// One wave32 handles a group of 32 points as two 16-point WMMA tiles (A and B),
// software-pipelined so each tile's repack VALU fills the other's WMMA hazard
// window. Each layer = one v_wmma_f32_16x16x32_f16 per tile; the K=16 padding
// slot carries the bias (A[M][16]=bias, B[16][n]=1.0). B slots for K=17..31 are
// don't-care (A columns there are exactly zero and all values are finite).
__global__ __launch_bounds__(256) void cppn_wmma_kernel(
    const float* __restrict__ x,   // [N,3]
    const float* __restrict__ W0,  // [16,3]
    const float* __restrict__ b0,  // [16]
    const float* __restrict__ Wh,  // [8,16,16]
    const float* __restrict__ bh,  // [8,16]
    const float* __restrict__ Wo,  // [1,16]
    const float* __restrict__ bo,  // [1]
    float* __restrict__ out,       // [N]
    int n, int ngroups)
{
    const int lane = threadIdx.x & 31;
    const int M    = lane & 15;        // A: row index; B: column (point) index
    const bool up  = lane >= 16;
    const int kb   = up ? 8 : 0;       // A f16 layout: lanes 16-31 hold K=8..15

    // ---------------- Preload A fragments (weights -> f16, bias in K=16) ----
    // A 16x32 f16 layout: lanes 0-15: V0..3 = K0..7, V4..7 = K16..23
    //                     lanes 16-31: V0..3 = K8..15, V4..7 = K24..31
    Frag a0 = {};
    if (!up) {
        a0.h[0] = pk(W0[M * 3 + 0], W0[M * 3 + 1]);
        a0.h[1] = pk(W0[M * 3 + 2], 0.0f);
        a0.h[4] = pk(b0[M], 0.0f);             // bias column K=16
    }
    Frag aW[8] = {};
#pragma unroll
    for (int l = 0; l < 8; ++l) {
        const float* w = Wh + l * 256 + M * 16 + kb;
        aW[l].h[0] = pk(w[0], w[1]);
        aW[l].h[1] = pk(w[2], w[3]);
        aW[l].h[2] = pk(w[4], w[5]);
        aW[l].h[3] = pk(w[6], w[7]);
        if (!up) aW[l].h[4] = pk(bh[l * 16 + M], 0.0f);   // bias column K=16
    }
    float wo[8];
#pragma unroll
    for (int r = 0; r < 8; ++r) wo[r] = Wo[r + (up ? 8 : 0)];
    const float bout = bo[0];

    const v2h one0  = {(_Float16)1.0f, (_Float16)0.0f};  // B bias row K=16 = 1.0
    const v8f czero = {};

    // tanh(D) -> next-layer B fragment (pack to f16 first, then swap 4 dwords)
    auto repack = [&](const v8f& d) -> Frag {
        v2h q[4], r[4];
#pragma unroll
        for (int v = 0; v < 4; ++v)
            q[v] = pk(fast_tanh(d[2 * v]), fast_tanh(d[2 * v + 1]));
#pragma unroll
        for (int v = 0; v < 4; ++v) r[v] = xh(q[v]);
        // Lower lanes: V0..3 = rows 0..7 (own, K0..7), V4 low = K=16 bias 1.0.
        // Upper lanes: V0..3 = rows 8..15 (own, K8..15).
        Frag nb;
        nb.h[0] = q[0];
        nb.h[1] = q[1]; nb.h[2] = q[2]; nb.h[3] = q[3];
        nb.h[4] = one0; nb.h[5] = r[1]; nb.h[6] = r[2]; nb.h[7] = r[3];
        return nb;
    };

    const int wid    = (blockIdx.x * blockDim.x + threadIdx.x) >> 5;
    const int nwaves = (gridDim.x * blockDim.x) >> 5;

    for (int g = wid; g < ngroups; g += nwaves) {
        // Each lane owns point g*32 + lane; lanes 0-15 -> tile A, 16-31 -> tile B.
        const long long pt  = (long long)g * 32 + lane;
        const long long ptc = pt < (long long)n ? pt : (long long)n - 1;
        const float* xp = x + ptc * 3;
        const v2h own0 = pk(xp[0], xp[1]);
        const v2h own1 = pk(xp[2], 0.0f);
        const v2h oth0 = xh(own0);            // partner half's point coords
        const v2h oth1 = xh(own1);

        // ---- Input-layer B frags (K=0..2 = x, K=16 = 1.0, rest don't-care) --
        Frag bA = {}, bB = {};
        bA.h[0] = own0;  bA.h[1] = own1;  bA.h[4] = one0;
        bB.h[0] = oth0;  bB.h[1] = oth1;  bB.h[4] = one0;

        v8f dA = __builtin_amdgcn_wmma_f32_16x16x32_f16(
            false, a0.v, false, bA.v, (short)0, czero, false, false);
        v8f dB = __builtin_amdgcn_wmma_f32_16x16x32_f16(
            false, a0.v, false, bB.v, (short)0, czero, false, false);

        // ---- 8 hidden layers, two tiles pipelined half a stage apart -------
#pragma unroll
        for (int l = 0; l < 8; ++l) {
            Frag nA = repack(dA);
            dA = __builtin_amdgcn_wmma_f32_16x16x32_f16(
                false, aW[l].v, false, nA.v, (short)0, czero, false, false);
            Frag nB = repack(dB);
            dB = __builtin_amdgcn_wmma_f32_16x16x32_f16(
                false, aW[l].v, false, nB.v, (short)0, czero, false, false);
        }

        // ---- Output layer: dot(tanh(h), Wo) + sigmoid ----------------------
        float pA = 0.0f, pB = 0.0f;
#pragma unroll
        for (int r = 0; r < 8; ++r) {
            pA += fast_tanh(dA[r]) * wo[r];
            pB += fast_tanh(dB[r]) * wo[r];
        }
        // Each lane stores the tile matching its half; exchange only the
        // other-tile partial so one swap completes both column sums.
        const float own   = up ? pB : pA;
        const float sent  = up ? pA : pB;
        const float total = own + xf(sent);
        const float o = fast_sigmoid(total + bout);
        if (pt < (long long)n) out[pt] = o;
    }
}

extern "C" void kernel_launch(void* const* d_in, const int* in_sizes, int n_in,
                              void* d_out, int out_size, void* d_ws, size_t ws_size,
                              hipStream_t stream) {
    const float* x  = (const float*)d_in[0];
    const float* W0 = (const float*)d_in[1];
    const float* b0 = (const float*)d_in[2];
    const float* Wh = (const float*)d_in[3];
    const float* bh = (const float*)d_in[4];
    const float* Wo = (const float*)d_in[5];
    const float* bo = (const float*)d_in[6];
    float* out = (float*)d_out;

    const int n       = in_sizes[0] / 3;       // number of points
    const int ngroups = (n + 31) / 32;         // 32 points (2 WMMA tiles) per group

    const int block = 256;                     // 8 waves per block
    int grid = 1024;                           // 8192 waves; grid-stride over groups
    const int max_grid = (ngroups + 7) / 8;    // >=1 group per wave
    if (grid > max_grid && max_grid > 0) grid = max_grid;

    cppn_wmma_kernel<<<grid, block, 0, stream>>>(x, W0, b0, Wh, bh, Wo, bo,
                                                 out, n, ngroups);
}
